// PTGAMiniExpert_38259568672952
// MI455X (gfx1250) — matrix operand order repaired and checked by reading, not compiled, not run on hardware
//
#include <hip/hip_runtime.h>
#include <math.h>

typedef __attribute__((ext_vector_type(16))) _Float16 v16h;
typedef __attribute__((ext_vector_type(8)))  float    v8f;

// ---------------------------------------------------------------------------
// Fragment packing: convert row-major f32 matrices into WMMA-fragment-ordered
// f16 so each GEMM operand load is one lane-contiguous 32-byte load.
// Layouts per CDNA5 ISA 7.12.2 (wave32).
// ---------------------------------------------------------------------------

// A [M,K] -> fragments (M/16) x (K/32); fragment f, lane l owns 16 halves at
// out[(f*32 + l)*16]. Lane l: m = l%16, half = l/16, VGPR v: k = (v<4?2v:2v+8)+8*half.
__global__ void pack_a_kernel(const float* __restrict__ A, int M, int K,
                              _Float16* __restrict__ out) {
  int gid = blockIdx.x * blockDim.x + threadIdx.x;
  int total = (M >> 4) * (K >> 5) * 32;
  if (gid >= total) return;
  int lane = gid & 31;
  int frag = gid >> 5;
  int KT = K >> 5;
  int rowTile = frag / KT;
  int kt = frag - rowTile * KT;
  int row0 = rowTile * 16, k0 = kt * 32;
  int m = lane & 15, hl = lane >> 4;
  v16h a;
#pragma unroll
  for (int v = 0; v < 8; ++v) {
    int k = (v < 4 ? 2 * v : 2 * v + 8) + 8 * hl;
    const float* p = A + (size_t)(row0 + m) * K + (k0 + k);
    a[2 * v]     = (_Float16)p[0];
    a[2 * v + 1] = (_Float16)p[1];
  }
  *(v16h*)(out + ((size_t)frag * 32 + lane) * 16) = a;
}

// B [K,N] -> fragments (K/32) x (N/16), frag index kt*(N/16)+nt.
// Lane l: n = l%16, half = l/16, VGPR v: k = (v<4?2v:2v+8)+8*half (+row pair).
__global__ void pack_b_kernel(const float* __restrict__ B, int K, int N,
                              _Float16* __restrict__ out) {
  int gid = blockIdx.x * blockDim.x + threadIdx.x;
  int total = (K >> 5) * (N >> 4) * 32;
  if (gid >= total) return;
  int lane = gid & 31;
  int frag = gid >> 5;
  int ntiles = N >> 4;
  int kt = frag / ntiles;
  int nt = frag - kt * ntiles;
  int n = lane & 15, hl = lane >> 4;
  v16h b;
#pragma unroll
  for (int v = 0; v < 8; ++v) {
    int k = (v < 4 ? 2 * v : 2 * v + 8) + 8 * hl;
    const float* p = B + (size_t)(kt * 32 + k) * N + (nt * 16 + n);
    b[2 * v]     = (_Float16)p[0];
    b[2 * v + 1] = (_Float16)p[N];
  }
  *(v16h*)(out + ((size_t)frag * 32 + lane) * 16) = b;
}

// ---------------------------------------------------------------------------
// GEMM on packed fragments: C[M,N] = A @ B (+bias). One wave: 16 x (NT*16).
// Software-pipelined: iteration kt+1's fragments are in flight while kt's
// WMMAs issue. Per kt: one A base + one B base, loads use immediate offsets.
// ---------------------------------------------------------------------------
template <int NT>
__global__ void gemm_wmma_kernel(const _Float16* __restrict__ Apk,
                                 const _Float16* __restrict__ Bpk,
                                 const float* __restrict__ bias, float* __restrict__ C,
                                 int M, int K, int N) {
  int wid = (blockIdx.x * blockDim.x + threadIdx.x) >> 5;
  int colGroups = N / (NT * 16);
  int rowTile = wid / colGroups;
  int cg = wid - rowTile * colGroups;
  if (rowTile * 16 >= M) return;
  int KT = K >> 5;
  int ntiles = N >> 4;
  int lane = threadIdx.x & 31;

  // lane-contiguous fragment pointers; advance by constant strides per kt
  const _Float16* aP = Apk + ((size_t)rowTile * KT * 32 + lane) * 16;
  const _Float16* bP = Bpk + ((size_t)(cg * NT) * 32 + lane) * 16;
  const size_t bStride = (size_t)ntiles * 512;  // halves per kt step

  v8f acc[NT];
  v8f zero = {};
#pragma unroll
  for (int t = 0; t < NT; ++t) acc[t] = zero;

  v16h a = *(const v16h*)aP;
  v16h b[NT];
#pragma unroll
  for (int t = 0; t < NT; ++t) b[t] = *(const v16h*)(bP + t * 512);

  for (int kt = 0; kt < KT - 1; ++kt) {
    aP += 512;
    bP += bStride;
    v16h an = *(const v16h*)aP;   // prefetch next iteration
    v16h bn[NT];
#pragma unroll
    for (int t = 0; t < NT; ++t) bn[t] = *(const v16h*)(bP + t * 512);
#pragma unroll
    for (int t = 0; t < NT; ++t)
      acc[t] = __builtin_amdgcn_wmma_f32_16x16x32_f16(false, a, false, b[t],
                                                      (short)0, acc[t], false, false);
    a = an;
#pragma unroll
    for (int t = 0; t < NT; ++t) b[t] = bn[t];
  }
#pragma unroll
  for (int t = 0; t < NT; ++t)
    acc[t] = __builtin_amdgcn_wmma_f32_16x16x32_f16(false, a, false, b[t],
                                                    (short)0, acc[t], false, false);

  // C/D layout: lane l, VGPR v -> row = v + 8*(l/16), col = l%16.
  int n = lane & 15;
  int hl = lane >> 4;
  int row0 = rowTile * 16;
#pragma unroll
  for (int t = 0; t < NT; ++t) {
    int col = (cg * NT + t) * 16 + n;
    float bv = bias ? bias[col] : 0.0f;
#pragma unroll
    for (int v = 0; v < 8; ++v) {
      int r = v + 8 * hl;
      C[(size_t)(row0 + r) * N + col] = acc[t][v] + bv;
    }
  }
}

// ---------------------------------------------------------------------------
// Helpers
// ---------------------------------------------------------------------------
__device__ __forceinline__ float warp_allreduce_sum(float v) {
#pragma unroll
  for (int off = 16; off >= 1; off >>= 1) v += __shfl_xor(v, off, 32);
  return v;
}

__device__ __forceinline__ void atomicMaxFloat(float* addr, float val) {
  int* ia = (int*)addr;
  int old = __float_as_int(*addr);
  while (val > __int_as_float(old)) {
    int assumed = old;
    old = atomicCAS(ia, assumed, __float_as_int(val));
    if (old == assumed) break;
  }
}

__global__ void fill_kernel(float* __restrict__ p, float v, int n) {
  int i = blockIdx.x * blockDim.x + threadIdx.x;
  if (i < n) p[i] = v;
}

// ---------------------------------------------------------------------------
// LayerNorm(row of 128) * gamma + beta, then * sigmoid(gate_logits[row%32]).
// ---------------------------------------------------------------------------
__global__ void ln_gate_kernel(const float* __restrict__ aligned,
                               const float* __restrict__ gamma,
                               const float* __restrict__ beta,
                               const float* __restrict__ gate_logits,
                               float* __restrict__ x0) {
  int row = (blockIdx.x * blockDim.x + threadIdx.x) >> 5;
  int lane = threadIdx.x & 31;
  const float* a = aligned + (size_t)row * 128 + lane * 4;
  float v0 = a[0], v1 = a[1], v2 = a[2], v3 = a[3];
  float s = warp_allreduce_sum(v0 + v1 + v2 + v3);
  float mu = s * (1.0f / 128.0f);
  float d0 = v0 - mu, d1 = v1 - mu, d2 = v2 - mu, d3 = v3 - mu;
  float sq = warp_allreduce_sum(d0 * d0 + d1 * d1 + d2 * d2 + d3 * d3);
  float rstd = rsqrtf(sq * (1.0f / 128.0f) + 1e-5f);
  float g = 1.0f / (1.0f + __expf(-gate_logits[row & 31]));
  const float* gm = gamma + lane * 4;
  const float* bt = beta + lane * 4;
  float* o = x0 + (size_t)row * 128 + lane * 4;
  o[0] = (d0 * rstd * gm[0] + bt[0]) * g;
  o[1] = (d1 * rstd * gm[1] + bt[1]) * g;
  o[2] = (d2 * rstd * gm[2] + bt[2]) * g;
  o[3] = (d3 * rstd * gm[3] + bt[3]) * g;
}

// ---------------------------------------------------------------------------
// Per-node attention logits: es[n,h] = dot(h[n,h,:], a_src[h,:]); same for ed.
// ---------------------------------------------------------------------------
__global__ void logits_kernel(const float* __restrict__ hfeat,
                              const float* __restrict__ a_src,
                              const float* __restrict__ a_dst,
                              float* __restrict__ es, float* __restrict__ ed, int H) {
  int node = (blockIdx.x * blockDim.x + threadIdx.x) >> 5;
  int lane = threadIdx.x & 31;
  int HF = H * 128;
  const float* hp = hfeat + (size_t)node * HF;
  for (int h = 0; h < H; ++h) {
    float ps = 0.0f, pd = 0.0f;
#pragma unroll
    for (int j = 0; j < 4; ++j) {
      int c = h * 128 + lane + 32 * j;
      float v = hp[c];
      ps += v * a_src[c];
      pd += v * a_dst[c];
    }
    ps = warp_allreduce_sum(ps);
    pd = warp_allreduce_sum(pd);
    if (lane == 0) {
      es[node * H + h] = ps;
      ed[node * H + h] = pd;
    }
  }
}

// ---------------------------------------------------------------------------
// Segment softmax over dst (3 passes: max, exp-sum, weighted aggregate).
// ---------------------------------------------------------------------------
__global__ void edge_max_kernel(const int* __restrict__ src, const int* __restrict__ dst,
                                const float* __restrict__ es, const float* __restrict__ ed,
                                float* __restrict__ m, int E, int H) {
  int e = blockIdx.x * blockDim.x + threadIdx.x;
  if (e >= E) return;
  int s = src[e], d = dst[e];
  for (int h = 0; h < H; ++h) {
    float l = es[s * H + h] + ed[d * H + h];
    l = l > 0.0f ? l : 0.2f * l;  // leaky_relu(0.2)
    atomicMaxFloat(&m[d * H + h], l);
  }
}

__global__ void edge_expsum_kernel(const int* __restrict__ src, const int* __restrict__ dst,
                                   const float* __restrict__ es, const float* __restrict__ ed,
                                   const float* __restrict__ m, float* __restrict__ den,
                                   int E, int H) {
  int e = blockIdx.x * blockDim.x + threadIdx.x;
  if (e >= E) return;
  int s = src[e], d = dst[e];
  for (int h = 0; h < H; ++h) {
    float l = es[s * H + h] + ed[d * H + h];
    l = l > 0.0f ? l : 0.2f * l;
    atomicAdd(&den[d * H + h], __expf(l - m[d * H + h]));
  }
}

// One block per edge, 128 threads over features; H heads of 128 features.
__global__ void edge_agg_kernel(const int* __restrict__ src, const int* __restrict__ dst,
                                const float* __restrict__ es, const float* __restrict__ ed,
                                const float* __restrict__ m, const float* __restrict__ den,
                                const float* __restrict__ hfeat, float* __restrict__ out,
                                int H) {
  int e = blockIdx.x;
  int tid = threadIdx.x;  // 0..127
  int s = src[e], d = dst[e];
  int HF = H * 128;
  for (int h = 0; h < H; ++h) {
    float l = es[s * H + h] + ed[d * H + h];
    l = l > 0.0f ? l : 0.2f * l;
    float alpha = __expf(l - m[d * H + h]) / den[d * H + h];
    int c = h * 128 + tid;
    atomicAdd(&out[(size_t)d * HF + c], alpha * hfeat[(size_t)s * HF + c]);
  }
}

// x = elu(x + bias[i % 512]), in place.
__global__ void elu_bias_kernel(float* __restrict__ x, const float* __restrict__ bias, int n) {
  int i = blockIdx.x * blockDim.x + threadIdx.x;
  if (i >= n) return;
  float v = x[i] + bias[i & 511];
  x[i] = v > 0.0f ? v : (__expf(v) - 1.0f);
}

// Mean over the 32 nodes of each graph, + bias2. grid=1024 blocks, 128 threads.
__global__ void pool_kernel(const float* __restrict__ out2, const float* __restrict__ bias2,
                            float* __restrict__ y) {
  int g = blockIdx.x, f = threadIdx.x;
  const float* p = out2 + (size_t)g * 32 * 128 + f;
  float s = 0.0f;
#pragma unroll
  for (int i = 0; i < 32; ++i) s += p[i * 128];
  y[(size_t)g * 128 + f] = s * (1.0f / 32.0f) + bias2[f];
}

__global__ void gate_kernel(const float* __restrict__ gl, float* __restrict__ y) {
  int i = threadIdx.x;
  y[i] = 1.0f / (1.0f + __expf(-gl[i]));
}

// ---------------------------------------------------------------------------
// Host launch
// ---------------------------------------------------------------------------
extern "C" void kernel_launch(void* const* d_in, const int* in_sizes, int n_in,
                              void* d_out, int out_size, void* d_ws, size_t ws_size,
                              hipStream_t stream) {
  const float* x_fields    = (const float*)d_in[0];
  const float* aligner_w   = (const float*)d_in[1];
  const float* aligner_b   = (const float*)d_in[2];
  const float* ln_gamma    = (const float*)d_in[3];
  const float* ln_beta     = (const float*)d_in[4];
  const float* gate_logits = (const float*)d_in[5];
  const float* w1          = (const float*)d_in[6];
  const float* att_src1    = (const float*)d_in[7];
  const float* att_dst1    = (const float*)d_in[8];
  const float* bias1       = (const float*)d_in[9];
  const float* w2          = (const float*)d_in[10];
  const float* att_src2    = (const float*)d_in[11];
  const float* att_dst2    = (const float*)d_in[12];
  const float* bias2       = (const float*)d_in[13];
  const int*   edge_index  = (const int*)d_in[14];
  // d_in[15] = batch_idx; layout is known (node/32), not needed.

  const int E = in_sizes[14] / 2;  // 294912
  const int* esrc = edge_index;
  const int* edst = edge_index + E;

  // Workspace layout (f32 units). Total ~55.25M floats = 221 MB.
  float* ws      = (float*)d_ws;
  float* aligned = ws;                  //  4,194,304 (reused as h2 later)
  float* x0      = ws + 4194304;        //  4,194,304
  float* h1      = ws + 8388608;        // 16,777,216
  float* out1    = ws + 25165824;       // 16,777,216 (becomes x1 after ELU)
  float* out2    = ws + 41943040;       //  4,194,304
  float* es1     = ws + 46137344;       //    131,072
  float* ed1     = es1 + 131072;
  float* m1      = ed1 + 131072;
  float* den1    = m1 + 131072;
  float* es2     = den1 + 131072;       //     32,768 each
  float* ed2     = es2 + 32768;
  float* m2      = ed2 + 32768;
  float* den2    = m2 + 32768;
  float* h2      = aligned;
  // packed f16 buffers (sized in halves)
  _Float16* Apk  = (_Float16*)(ws + 46792704);  // up to 16,777,216 halves (reused per GEMM)
  _Float16* Bpk1 = (_Float16*)(ws + 55181312);  //   8,192 halves
  _Float16* Bpk2 = (_Float16*)(ws + 55185408);  //  65,536 halves
  _Float16* Bpk3 = (_Float16*)(ws + 55218176);  //  65,536 halves

  const float NEG_BIG = -1.0e30f;
  int eb = (E + 255) / 256;

  // 0) pack the three weight matrices (tiny, once per call)
  pack_b_kernel<<<2, 256, 0, stream>>>(aligner_w, 64, 128, Bpk1);    //   512 threads
  pack_b_kernel<<<16, 256, 0, stream>>>(w1, 128, 512, Bpk2);         //  4096 threads
  pack_b_kernel<<<16, 256, 0, stream>>>(w2, 512, 128, Bpk3);         //  4096 threads

  // 1) aligned = x_fields @ aligner_w + aligner_b   [32768,64]@[64,128]
  pack_a_kernel<<<512, 256, 0, stream>>>(x_fields, 32768, 64, Apk);
  gemm_wmma_kernel<4><<<512, 256, 0, stream>>>(Apk, Bpk1, aligner_b, aligned,
                                               32768, 64, 128);
  // 2) LayerNorm * gamma + beta, * gate -> x0
  ln_gate_kernel<<<4096, 256, 0, stream>>>(aligned, ln_gamma, ln_beta, gate_logits, x0);

  // 3) h1 = x0 @ w1   [32768,128]@[128,512]
  pack_a_kernel<<<1024, 256, 0, stream>>>(x0, 32768, 128, Apk);
  gemm_wmma_kernel<4><<<2048, 256, 0, stream>>>(Apk, Bpk2, nullptr, h1,
                                                32768, 128, 512);

  // 4) GAT layer 1 segment softmax + aggregate
  logits_kernel<<<4096, 256, 0, stream>>>(h1, att_src1, att_dst1, es1, ed1, 4);
  fill_kernel<<<512, 256, 0, stream>>>(m1, NEG_BIG, 131072);
  fill_kernel<<<512, 256, 0, stream>>>(den1, 0.0f, 131072);
  fill_kernel<<<65536, 256, 0, stream>>>(out1, 0.0f, 16777216);
  edge_max_kernel<<<eb, 256, 0, stream>>>(esrc, edst, es1, ed1, m1, E, 4);
  edge_expsum_kernel<<<eb, 256, 0, stream>>>(esrc, edst, es1, ed1, m1, den1, E, 4);
  edge_agg_kernel<<<E, 128, 0, stream>>>(esrc, edst, es1, ed1, m1, den1, h1, out1, 4);
  // 5) x1 = elu(out1 + bias1), in place
  elu_bias_kernel<<<65536, 256, 0, stream>>>(out1, bias1, 16777216);

  // 6) h2 = x1 @ w2   [32768,512]@[512,128]
  pack_a_kernel<<<4096, 256, 0, stream>>>(out1, 32768, 512, Apk);
  gemm_wmma_kernel<4><<<512, 256, 0, stream>>>(Apk, Bpk3, nullptr, h2,
                                               32768, 512, 128);

  // 7) GAT layer 2 (1 head)
  logits_kernel<<<4096, 256, 0, stream>>>(h2, att_src2, att_dst2, es2, ed2, 1);
  fill_kernel<<<128, 256, 0, stream>>>(m2, NEG_BIG, 32768);
  fill_kernel<<<128, 256, 0, stream>>>(den2, 0.0f, 32768);
  fill_kernel<<<16384, 256, 0, stream>>>(out2, 0.0f, 4194304);
  edge_max_kernel<<<eb, 256, 0, stream>>>(esrc, edst, es2, ed2, m2, E, 1);
  edge_expsum_kernel<<<eb, 256, 0, stream>>>(esrc, edst, es2, ed2, m2, den2, E, 1);
  edge_agg_kernel<<<E, 128, 0, stream>>>(esrc, edst, es2, ed2, m2, den2, h2, out2, 1);

  // 8) graph mean pool (+bias2) -> d_out[0:131072]; gate -> d_out[131072:131104]
  float* y = (float*)d_out;
  pool_kernel<<<1024, 128, 0, stream>>>(out2, bias2, y);
  gate_kernel<<<1, 32, 0, stream>>>(gate_logits, y + 131072);
}
